// MMRI_P2I_33844342292491
// MI455X (gfx1250) — compile-verified, matrix operand order checked by reading, not run on hardware
//
#include <hip/hip_runtime.h>
#include <hip/hip_bf16.h>

// ---------------- problem constants ----------------
#define BN    12
#define CCH   256
#define HH    40
#define WW    100
#define NQ    (HH * WW)          // 4000
#define MROWS (BN * NQ)          // 48000
#define HEADS 8
#define PTS   4
#define HD    (CCH / HEADS)      // 32
#define HP2   (HEADS * PTS * 2)  // 64
#define HP    (HEADS * PTS)      // 32

// GEMM tiling
#define MB_T  128                // block tile M
#define NB_T  64                 // block tile N
#define KS    32                 // K step (one WMMA K)

typedef __attribute__((ext_vector_type(16))) _Float16     v16h;
typedef __attribute__((ext_vector_type(8)))  float        v8f;
typedef __attribute__((ext_vector_type(4)))  unsigned int u32x4;
typedef __attribute__((ext_vector_type(4)))  int          v4i;

#define AS1 __attribute__((address_space(1)))
#define AS3 __attribute__((address_space(3)))

#if defined(__has_builtin)
#  if __has_builtin(__builtin_amdgcn_global_load_async_to_lds_b128) && \
      __has_builtin(__builtin_amdgcn_s_wait_asynccnt)
#    define USE_ASYNC_LDS 1
#  else
#    define USE_ASYNC_LDS 0
#  endif
#else
#  define USE_ASYNC_LDS 0
#endif

// ---------------------------------------------------------------------------
// Weight transpose + f32 -> f16 convert:  Wdst[n*K + k] = (f16) Wsrc[k*N + n]
// ---------------------------------------------------------------------------
__global__ void k_transpose16(const float* __restrict__ Wsrc,
                              _Float16* __restrict__ Wdst, int K, int N) {
    int idx = blockIdx.x * blockDim.x + threadIdx.x;
    if (idx >= K * N) return;
    int k = idx / N, n = idx - k * N;
    Wdst[(size_t)n * K + k] = (_Float16)Wsrc[(size_t)k * N + n];
}

// ---------------------------------------------------------------------------
// BEV warp: bilinear sample img_feats at warp_grid -> warped (f16),
// and pack query = transposed img_feats (f16). Block = 256 threads = channels,
// one block per (b, q).
// ---------------------------------------------------------------------------
__global__ void k_warp(const float* __restrict__ img,       // [BN,C,H,W]
                       const float* __restrict__ grid,      // [BN,nq,2]
                       _Float16* __restrict__ warped16,     // [M,C]
                       _Float16* __restrict__ query16) {    // [M,C]
    int bq = blockIdx.x;                 // 0..M-1
    int b  = bq / NQ;
    int q  = bq - b * NQ;
    int c  = threadIdx.x;

    float gx = grid[(size_t)bq * 2 + 0] * (float)WW - 0.5f;
    float gy = grid[(size_t)bq * 2 + 1] * (float)HH - 0.5f;
    float x0f = floorf(gx), y0f = floorf(gy);
    float wx = gx - x0f, wy = gy - y0f;
    int x0 = (int)x0f, y0 = (int)y0f;

    const float* plane = img + ((size_t)b * CCH + c) * (size_t)NQ;

    float acc = 0.f;
#pragma unroll
    for (int dy = 0; dy < 2; ++dy) {
#pragma unroll
        for (int dx = 0; dx < 2; ++dx) {
            int xi = x0 + dx, yi = y0 + dy;
            float w = (dx ? wx : 1.f - wx) * (dy ? wy : 1.f - wy);
            if (xi >= 0 && xi < WW && yi >= 0 && yi < HH)
                acc += w * plane[yi * WW + xi];
        }
    }
    warped16[(size_t)bq * CCH + c] = (_Float16)acc;
    query16 [(size_t)bq * CCH + c] = (_Float16)plane[q];
}

// ---------------------------------------------------------------------------
// Double-buffered LDS-staged WMMA GEMM:
//   out[M,N] = A[M,K](f16) * Bt[N,K]^T(f16) + bias[N]
//
// Block = 256 threads (8 waves). Block tile = 128(M) x 64(N), K stepped by 32.
// Two LDS buffers (24 KB total): while computing K-step s from buf[s&1], the
// async copies for K-step s+1 are already in flight into buf[(s+1)&1];
// s_wait_asynccnt(3) fences only the older (in-order) 3 loads of the current
// buffer, overlapping global latency with WMMA. Wave w computes rows
// [w*16, w*16+16) x 64 cols = 4 WMMA tiles; B fragments shared via LDS.
//
// mode 0: plain row-major f32 store
// mode 2: fused final store: out[(b*N+n)*nq+q] = acc + bias[n] + resid[same]
//         (output transpose + residual; resid = img_feats)
// Requires M % 128 == 0; N handled via clamp/guard (N in {32, 64, 256}).
// ---------------------------------------------------------------------------
__global__ void __launch_bounds__(256)
k_gemm_wmma(const _Float16* __restrict__ A,
            const _Float16* __restrict__ Bt,
            const float* __restrict__ bias,
            float* __restrict__ out,
            const float* __restrict__ resid,
            int M, int N, int K, int nBlocks, int mode) {
    __shared__ _Float16 sA[2][MB_T * KS];   // 2 x 8 KB
    __shared__ _Float16 sB[2][NB_T * KS];   // 2 x 4 KB

    int tid = threadIdx.x;
    int mb  = blockIdx.x / nBlocks;
    int nb  = blockIdx.x - mb * nBlocks;
    int mBase = mb * MB_T;
    int nBase = nb * NB_T;

    int w    = tid >> 5;      // wave id 0..7
    int l    = tid & 31;
    int half = l >> 4;        // K-half selector per ISA 16-bit A layout
    int mr   = l & 15;        // row (A) / col (B,D) within 16

    // ---- staging assignment: 16B chunks ----
    // A tile: 128 rows x 64B = 512 chunks -> each thread does rows rA, rA+64
    // B tile:  64 rows x 64B = 256 chunks -> each thread does row rB
    int rA = tid >> 2;
    int cc = tid & 3;                       // 16B chunk within row
    int rB = rA;
    int rBg = nBase + rB;
    if (rBg > N - 1) rBg = N - 1;           // clamp (N=32 case)

    const _Float16* gA0 = A  + (size_t)(mBase + rA) * K + cc * 8;
    const _Float16* gB0 = Bt + (size_t)rBg * K + cc * 8;
    const int lOffA = rA * KS + cc * 8;
    const int lOffB = rB * KS + cc * 8;

    union Frag { v16h v; u32x4 q[2]; };
    v8f acc0 = {}, acc1 = {}, acc2 = {}, acc3 = {};

    const int aRow = w * 16 + mr;           // this lane's A row in block tile

    auto issueLoads = [&](int k0, int buf) {
#if USE_ASYNC_LDS
        __builtin_amdgcn_global_load_async_to_lds_b128(
            (AS1 v4i*)(gA0 + k0), (AS3 v4i*)(&sA[buf][lOffA]), 0, 0);
        __builtin_amdgcn_global_load_async_to_lds_b128(
            (AS1 v4i*)(gA0 + (size_t)64 * K + k0),
            (AS3 v4i*)(&sA[buf][lOffA + 64 * KS]), 0, 0);
        __builtin_amdgcn_global_load_async_to_lds_b128(
            (AS1 v4i*)(gB0 + k0), (AS3 v4i*)(&sB[buf][lOffB]), 0, 0);
#else
        u32x4 va0 = *(const u32x4*)(gA0 + k0);
        u32x4 va1 = *(const u32x4*)(gA0 + (size_t)64 * K + k0);
        u32x4 vb0 = *(const u32x4*)(gB0 + k0);
        *(u32x4*)(&sA[buf][lOffA])            = va0;
        *(u32x4*)(&sA[buf][lOffA + 64 * KS])  = va1;
        *(u32x4*)(&sB[buf][lOffB])            = vb0;
#endif
    };

    auto computeStep = [&](int buf) {
        const _Float16* cA = sA[buf];
        const _Float16* cB = sB[buf];
        Frag a;                               // ds_load_b128 x2
        a.q[0] = *(const u32x4*)(cA + aRow * KS + half * 8);
        a.q[1] = *(const u32x4*)(cA + aRow * KS + 16 + half * 8);

        Frag b0, b1, b2, b3;                  // ds_load_b128 x8, shared tiles
        b0.q[0] = *(const u32x4*)(cB + (0 * 16 + mr) * KS + half * 8);
        b0.q[1] = *(const u32x4*)(cB + (0 * 16 + mr) * KS + 16 + half * 8);
        b1.q[0] = *(const u32x4*)(cB + (1 * 16 + mr) * KS + half * 8);
        b1.q[1] = *(const u32x4*)(cB + (1 * 16 + mr) * KS + 16 + half * 8);
        b2.q[0] = *(const u32x4*)(cB + (2 * 16 + mr) * KS + half * 8);
        b2.q[1] = *(const u32x4*)(cB + (2 * 16 + mr) * KS + 16 + half * 8);
        b3.q[0] = *(const u32x4*)(cB + (3 * 16 + mr) * KS + half * 8);
        b3.q[1] = *(const u32x4*)(cB + (3 * 16 + mr) * KS + 16 + half * 8);

        acc0 = __builtin_amdgcn_wmma_f32_16x16x32_f16(false, a.v, false, b0.v,
                                                      (short)0, acc0, false, false);
        acc1 = __builtin_amdgcn_wmma_f32_16x16x32_f16(false, a.v, false, b1.v,
                                                      (short)0, acc1, false, false);
        acc2 = __builtin_amdgcn_wmma_f32_16x16x32_f16(false, a.v, false, b2.v,
                                                      (short)0, acc2, false, false);
        acc3 = __builtin_amdgcn_wmma_f32_16x16x32_f16(false, a.v, false, b3.v,
                                                      (short)0, acc3, false, false);
    };

    const int nsteps = K / KS;              // 8 for K=256
    issueLoads(0, 0);                       // prologue: buffer 0 in flight

    for (int s = 0; s < nsteps - 1; ++s) {
        issueLoads((s + 1) * KS, (s + 1) & 1);   // next tile in flight
#if USE_ASYNC_LDS
        __builtin_amdgcn_s_wait_asynccnt(3);     // fence current buffer only
#endif
        __syncthreads();                         // current buffer visible to all
        computeStep(s & 1);
        __syncthreads();                         // all done reading before reuse
    }
#if USE_ASYNC_LDS
    __builtin_amdgcn_s_wait_asynccnt(0);
#endif
    __syncthreads();
    computeStep((nsteps - 1) & 1);

    // ---- store 4 fragments (16 rows x 64 cols per wave) ----
    v8f accs[4] = {acc0, acc1, acc2, acc3};
#pragma unroll
    for (int j = 0; j < 4; ++j) {
        int n = nBase + j * 16 + mr;
        if (n >= N) continue;
        float bn = bias ? bias[n] : 0.f;
#pragma unroll
        for (int r = 0; r < 8; ++r) {
            int m = mBase + w * 16 + half * 8 + r;
            if (mode == 0) {
                out[(size_t)m * N + n] = accs[j][r] + bn;
            } else {
                int b = m / NQ;
                int q = m - b * NQ;
                size_t idx = ((size_t)b * N + n) * (size_t)NQ + q;
                out[idx] = accs[j][r] + bn + resid[idx];
            }
        }
    }
}

// ---------------------------------------------------------------------------
// Softmax over pts(4) per (row, head), in place on attn logits [M, 32]
// ---------------------------------------------------------------------------
__global__ void k_softmax(float* __restrict__ attn) {
    int idx = blockIdx.x * blockDim.x + threadIdx.x;
    if (idx >= MROWS * HEADS) return;
    int m = idx / HEADS, h = idx - m * HEADS;
    float* p = attn + (size_t)m * HP + h * PTS;
    float a0 = p[0], a1 = p[1], a2 = p[2], a3 = p[3];
    float mx = fmaxf(fmaxf(a0, a1), fmaxf(a2, a3));
    float e0 = __expf(a0 - mx), e1 = __expf(a1 - mx);
    float e2 = __expf(a2 - mx), e3 = __expf(a3 - mx);
    float inv = 1.f / (e0 + e1 + e2 + e3);
    p[0] = e0 * inv; p[1] = e1 * inv; p[2] = e2 * inv; p[3] = e3 * inv;
}

// ---------------------------------------------------------------------------
// Deformable sampling + attention aggregation.
// One wave per (b,q,head); lane = channel d (0..31) -> coalesced gathers.
// agg16[m*C + h*32 + d] = sum_p attn * bilinear(v)
// ---------------------------------------------------------------------------
__global__ void k_sample(const float* __restrict__ vbuf,    // [M, C] f32
                         const float* __restrict__ offbuf,  // [M, 64]
                         const float* __restrict__ attnbuf, // [M, 32]
                         const float* __restrict__ refpts,  // [M, 2]
                         _Float16* __restrict__ agg16) {    // [M, C]
    int wid = blockIdx.x * 8 + (threadIdx.x >> 5);
    if (wid >= MROWS * HEADS) return;
    int d = threadIdx.x & 31;
    int m = wid / HEADS;
    int h = wid - m * HEADS;
    int b = m / NQ;

    float rx = refpts[(size_t)m * 2 + 0] * (float)WW - 0.5f;
    float ry = refpts[(size_t)m * 2 + 1] * (float)HH - 0.5f;

    const float* vb = vbuf + (size_t)b * NQ * CCH + h * HD + d;

    float acc = 0.f;
#pragma unroll
    for (int p = 0; p < PTS; ++p) {
        float lx = rx + offbuf[(size_t)m * HP2 + h * (PTS * 2) + p * 2 + 0];
        float ly = ry + offbuf[(size_t)m * HP2 + h * (PTS * 2) + p * 2 + 1];
        float aw = attnbuf[(size_t)m * HP + h * PTS + p];

        float x0f = floorf(lx), y0f = floorf(ly);
        float wx = lx - x0f, wy = ly - y0f;
        int x0 = (int)x0f, y0 = (int)y0f;

        float s = 0.f;
#pragma unroll
        for (int dy = 0; dy < 2; ++dy) {
#pragma unroll
            for (int dx = 0; dx < 2; ++dx) {
                int xi = x0 + dx, yi = y0 + dy;
                float w = (dx ? wx : 1.f - wx) * (dy ? wy : 1.f - wy);
                if (xi >= 0 && xi < WW && yi >= 0 && yi < HH)
                    s += w * vb[(size_t)(yi * WW + xi) * CCH];
            }
        }
        acc += aw * s;
    }
    agg16[(size_t)m * CCH + h * HD + d] = (_Float16)acc;
}

// ---------------------------------------------------------------------------
extern "C" void kernel_launch(void* const* d_in, const int* in_sizes, int n_in,
                              void* d_out, int out_size, void* d_ws, size_t ws_size,
                              hipStream_t stream) {
    const float* img   = (const float*)d_in[0];   // [BN,C,H,W]
    const float* grid  = (const float*)d_in[1];   // [BN,H,W,2]
    const float* refp  = (const float*)d_in[2];   // [BN,nq,1,2]
    const float* Wv    = (const float*)d_in[3];
    const float* bv    = (const float*)d_in[4];
    const float* Woff  = (const float*)d_in[5];
    const float* boff  = (const float*)d_in[6];
    const float* Wattn = (const float*)d_in[7];
    const float* battn = (const float*)d_in[8];
    const float* Wout  = (const float*)d_in[9];
    const float* bout  = (const float*)d_in[10];
    float* out = (float*)d_out;

    // workspace carve-up (256B aligned)
    char* ws = (char*)d_ws;
    size_t off = 0;
    auto carve = [&](size_t bytes) -> char* {
        char* p = ws + off;
        off = (off + bytes + 255) & ~(size_t)255;
        return p;
    };
    _Float16* q16     = (_Float16*)carve((size_t)MROWS * CCH * 2);
    _Float16* w16     = (_Float16*)carve((size_t)MROWS * CCH * 2);
    _Float16* agg16   = (_Float16*)carve((size_t)MROWS * CCH * 2);
    _Float16* WvT     = (_Float16*)carve((size_t)CCH * CCH * 2);
    _Float16* WoffT   = (_Float16*)carve((size_t)HP2 * CCH * 2);
    _Float16* WattnT  = (_Float16*)carve((size_t)HP  * CCH * 2);
    _Float16* WoutT   = (_Float16*)carve((size_t)CCH * CCH * 2);
    float*    vbuf    = (float*)carve((size_t)MROWS * CCH * 4);
    float*    offbuf  = (float*)carve((size_t)MROWS * HP2 * 4);
    float*    attnbuf = (float*)carve((size_t)MROWS * HP  * 4);
    (void)ws_size; (void)in_sizes; (void)n_in; (void)out_size;

    // 1. weight transpose + f16 convert
    {
        int t = CCH * CCH;
        k_transpose16<<<(t + 255) / 256, 256, 0, stream>>>(Wv,    WvT,    CCH, CCH);
        k_transpose16<<<(t + 255) / 256, 256, 0, stream>>>(Wout,  WoutT,  CCH, CCH);
        int t2 = CCH * HP2;
        k_transpose16<<<(t2 + 255) / 256, 256, 0, stream>>>(Woff,  WoffT,  CCH, HP2);
        int t3 = CCH * HP;
        k_transpose16<<<(t3 + 255) / 256, 256, 0, stream>>>(Wattn, WattnT, CCH, HP);
    }

    // 2. BEV warp + query pack
    k_warp<<<MROWS, CCH, 0, stream>>>(img, grid, w16, q16);

    // 3. GEMMs (double-buffered LDS-staged WMMA f16 -> f32)
    int mBlk = MROWS / MB_T;                       // 375
    auto nBlk = [](int N) { return (N + NB_T - 1) / NB_T; };

    // v = warped @ Wv + bv          [48000 x 256]
    k_gemm_wmma<<<mBlk * nBlk(CCH), 256, 0, stream>>>(
        w16, WvT, bv, vbuf, nullptr, MROWS, CCH, CCH, nBlk(CCH), 0);
    // off = query @ Woff + boff     [48000 x 64]
    k_gemm_wmma<<<mBlk * nBlk(HP2), 256, 0, stream>>>(
        q16, WoffT, boff, offbuf, nullptr, MROWS, HP2, CCH, nBlk(HP2), 0);
    // attn logits = query @ Wattn + battn   [48000 x 32]
    k_gemm_wmma<<<mBlk * nBlk(HP), 256, 0, stream>>>(
        q16, WattnT, battn, attnbuf, nullptr, MROWS, HP, CCH, nBlk(HP), 0);

    // 4. softmax over points
    k_softmax<<<(MROWS * HEADS + 255) / 256, 256, 0, stream>>>(attnbuf);

    // 5. deformable sampling + aggregation
    k_sample<<<(MROWS * HEADS + 7) / 8, 256, 0, stream>>>(
        vbuf, offbuf, attnbuf, refp, agg16);

    // 6. out = agg @ Wout + bout + query, stored transposed [BN, C, nq]
    //    residual in transposed layout == img_feats itself.
    k_gemm_wmma<<<mBlk * nBlk(CCH), 256, 0, stream>>>(
        agg16, WoutT, bout, out, img, MROWS, CCH, CCH, nBlk(CCH), 2);
}
